// GCN_22213570855120
// MI455X (gfx1250) — matrix operand order repaired and checked by reading, not compiled, not run on hardware
//
#include <hip/hip_runtime.h>

typedef __attribute__((ext_vector_type(16))) _Float16 v16h;
typedef __attribute__((ext_vector_type(8)))  float    v8f;
typedef __attribute__((ext_vector_type(4)))  int      v4i;

#define NN      50000
#define NE      800000
#define NG      64
#define INDIM   128
#define HID4    256
#define OUTDIM  256
#define LNEPS   1e-5f

// ---- probe for gfx1250 async global->LDS copy builtin (compile-safe) ----
#if defined(__has_builtin)
#  if __has_builtin(__builtin_amdgcn_global_load_async_to_lds_b128)
#    define HAVE_ASYNC_LDS 1
#  endif
#endif
#ifndef HAVE_ASYNC_LDS
#  define HAVE_ASYNC_LDS 0
#endif

#if HAVE_ASYNC_LDS
typedef __attribute__((address_space(1))) v4i GLB_v4i;   // global int4
typedef __attribute__((address_space(3))) v4i LDS_v4i;   // LDS int4
#endif

// ---------------------------------------------------------------- utilities
__global__ void zero_kernel(float* __restrict__ p, long n) {
    long i = (long)blockIdx.x * blockDim.x + threadIdx.x;
    if (i < n) p[i] = 0.f;
}

__global__ void degree_kernel(const int* __restrict__ src, const int* __restrict__ dst,
                              float* __restrict__ outdeg, float* __restrict__ indeg, int E) {
    int e = blockIdx.x * blockDim.x + threadIdx.x;
    if (e < E) {
        atomicAdd(&outdeg[src[e]], 1.f);
        atomicAdd(&indeg[dst[e]], 1.f);
    }
}

__global__ void norm_kernel(float* __restrict__ outn, float* __restrict__ inn, int n) {
    int i = blockIdx.x * blockDim.x + threadIdx.x;
    if (i < n) {
        outn[i] = rsqrtf(fmaxf(outn[i], 1.f));
        inn[i]  = rsqrtf(fmaxf(inn[i], 1.f));
    }
}

// W is K x N fp32 row-major; Bt is N x K f16 (transposed) so a WMMA B-fragment
// lane reads 16 contiguous halves (32B) per k-step.
__global__ void convert_weight_t(const float* __restrict__ W, _Float16* __restrict__ Bt,
                                 int K, int N) {
    int i = blockIdx.x * blockDim.x + threadIdx.x;
    if (i < K * N) {
        int k = i / N, n = i % N;
        Bt[(size_t)n * K + k] = (_Float16)W[i];
    }
}

// ------------------------------------------------ edge-weighted SpMM scatter
__global__ void spmm_kernel(const float* __restrict__ H, int F,
                            const int* __restrict__ src, const int* __restrict__ dst,
                            const float* __restrict__ w,
                            const float* __restrict__ srcScale,   // out_norm or null
                            float* __restrict__ out, int E) {
    int wv   = (int)((blockIdx.x * (long)blockDim.x + threadIdx.x) >> 5);
    int lane = threadIdx.x & 31;
    if (wv >= E) return;
    int s = src[wv], d = dst[wv];
    float c = w[wv];
    if (srcScale) c *= srcScale[s];
    const float* hs = H + (size_t)s * F;
    float*       od = out + (size_t)d * F;
    for (int f = lane; f < F; f += 32)
        atomicAdd(&od[f], c * hs[f]);
}

// ------------------------------------------------------------- WMMA GEMM
// C[M,N] = epi( (A * rowScaleA) @ B ), A fp32 (ldA == K, tile contiguous),
// Bt = B transposed f16 (N x K). One wave owns a 16x16 C tile.
// M%16==0, N%16==0, K%32==0. Block: 8 waves, same mtile, 8 consecutive ntiles.
template<bool SCALEA, bool SCALEC, bool RELU>
__global__ __launch_bounds__(256)
void gemm_wmma_f16(const float* __restrict__ A, const _Float16* __restrict__ Bt,
                   float* __restrict__ C, int K, int N, int ldC,
                   const float* __restrict__ rowScaleA,
                   const float* __restrict__ rowScaleC) {
    int waveInBlk = threadIdx.x >> 5;
    int lane      = threadIdx.x & 31;
    int mtile     = blockIdx.x;
    int ntile     = blockIdx.y * 8 + waveInBlk;

    int half = lane >> 4;                  // 0: lanes 0-15, 1: lanes 16-31
    int mrow = mtile * 16 + (lane & 15);
    int bcol = ntile * 16 + (lane & 15);

#if HAVE_ASYNC_LDS
    // Stage the whole contiguous 16xK fp32 A tile into LDS with async copies
    // (ASYNCcnt-tracked), shared by all 8 waves in the block.
    __shared__ float aTile[16 * 512];      // max K = 512 -> 32KB
    {
        const char* gsrc = (const char*)(A + (size_t)mtile * 16 * K);
        int bytes = 16 * K * 4;
        for (int off = threadIdx.x * 16; off < bytes; off += 256 * 16) {
            GLB_v4i* g = (GLB_v4i*)(uintptr_t)(gsrc + off);
            LDS_v4i* l = (LDS_v4i*)(unsigned)(uintptr_t)((char*)aTile + off);
            __builtin_amdgcn_global_load_async_to_lds_b128(g, l, 0, 0);
        }
#  if defined(__has_builtin) && __has_builtin(__builtin_amdgcn_s_wait_asynccnt)
        __builtin_amdgcn_s_wait_asynccnt(0);
#  else
        asm volatile("s_wait_asynccnt 0x0" ::: "memory");
#  endif
        __syncthreads();
    }
    const float* Arow = &aTile[(lane & 15) * K];
#else
    const float* Arow = A + (size_t)mrow * K;
#endif

    float ascale = SCALEA ? rowScaleA[mrow] : 1.f;
    const _Float16* Bcol = Bt + (size_t)bcol * K;

    v8f acc = {};
    for (int kk = 0; kk < K; kk += 32) {
        // A 16x32 fragment: half 0 -> K 0-15, half 1 -> K 16-31
        v16h a;
        int abase = kk + half * 16;
        #pragma unroll
        for (int i = 0; i < 8; ++i) {
            a[i]     = (_Float16)(Arow[abase + i]      * ascale);
            a[i + 8] = (_Float16)(Arow[abase + 8 + i]  * ascale);
        }
        // B 32x16 fragment: lane = column; half 0 -> K 0-15, half 1 -> K 16-31.
        // Transposed f16 layout -> one 32B contiguous load.
        v16h b = *(const v16h*)(Bcol + kk + half * 16);

        acc = __builtin_amdgcn_wmma_f32_16x16x32_f16(
                  false, a, false, b, (short)0, acc, false, false);
    }

    // C/D layout: lane element r -> (row = 16*mtile + r + 8*half, col = bcol)
    #pragma unroll
    for (int r = 0; r < 8; ++r) {
        int row = mtile * 16 + r + half * 8;
        float v = acc[r];
        if (SCALEC) v *= rowScaleC[row];
        if (RELU)   v = fmaxf(v, 0.f);
        C[(size_t)row * ldC + bcol] = v;
    }
}

// --------------------------------------------------- LayerNorm + ReLU (wave/row)
__global__ void layernorm_relu_kernel(float* __restrict__ H, int ld,
                                      const float* __restrict__ gamma,
                                      const float* __restrict__ beta, int M) {
    int wv   = (int)((blockIdx.x * (long)blockDim.x + threadIdx.x) >> 5);
    int lane = threadIdx.x & 31;
    if (wv >= M) return;
    float* row = H + (size_t)wv * ld;
    float vals[8];
    float s = 0.f, s2 = 0.f;
    #pragma unroll
    for (int i = 0; i < 8; ++i) {
        float v = row[lane + 32 * i];
        vals[i] = v; s += v; s2 += v * v;
    }
    #pragma unroll
    for (int off = 16; off > 0; off >>= 1) {
        s  += __shfl_xor(s, off, 32);
        s2 += __shfl_xor(s2, off, 32);
    }
    float mean = s * (1.f / HID4);
    float var  = s2 * (1.f / HID4) - mean * mean;
    float inv  = rsqrtf(var + LNEPS);
    #pragma unroll
    for (int i = 0; i < 8; ++i) {
        int c = lane + 32 * i;
        float y = (vals[i] - mean) * inv * gamma[c] + beta[c];
        row[c] = fmaxf(y, 0.f);
    }
}

// ------------------------------------------------------- row-scale + ReLU
__global__ void rowscale_relu_kernel(float* __restrict__ H,
                                     const float* __restrict__ scale, long M, int F) {
    long i = (long)blockIdx.x * blockDim.x + threadIdx.x;
    if (i < M * F) {
        long row = i / F;
        H[i] = fmaxf(H[i] * scale[row], 0.f);
    }
}

// ------------------------------------- per-graph readout (graph_ids sorted)
__global__ void readout_kernel(const float* __restrict__ X, const int* __restrict__ gid,
                               float* __restrict__ out, int M) {
    int c  = threadIdx.x;                 // 256 threads = 256 feature columns
    int n0 = blockIdx.x * 256;
    int n1 = n0 + 256; if (n1 > M) n1 = M;
    float acc = 0.f;
    int curg = gid[n0];
    for (int n = n0; n < n1; ++n) {
        int g = gid[n];
        if (g != curg) {
            atomicAdd(&out[(size_t)curg * OUTDIM + c], acc);
            acc = 0.f; curg = g;
        }
        acc += X[(size_t)n * OUTDIM + c];
    }
    atomicAdd(&out[(size_t)curg * OUTDIM + c], acc);
}

// ----------------------------------------------------------------- driver
extern "C" void kernel_launch(void* const* d_in, const int* in_sizes, int n_in,
                              void* d_out, int out_size, void* d_ws, size_t ws_size,
                              hipStream_t stream) {
    const float* x     = (const float*)d_in[0];
    const float* w     = (const float*)d_in[1];
    const float* W1    = (const float*)d_in[2];
    const float* Wfc   = (const float*)d_in[3];
    const float* gamma = (const float*)d_in[4];
    const float* beta  = (const float*)d_in[5];
    const float* W2    = (const float*)d_in[6];
    const float* W3    = (const float*)d_in[7];
    const int*   src   = (const int*)d_in[8];
    const int*   dst   = (const int*)d_in[9];
    const int*   gid   = (const int*)d_in[10];
    float*       out   = (float*)d_out;

    // workspace: norms | x1f1 (Nx512) | bufB (Nx256) | bufC (Nx256) | f16 weights
    float* outn = (float*)d_ws;
    float* inn  = outn + NN;
    float* x1f1 = inn + NN;
    float* bufB = x1f1 + (size_t)NN * 512;
    float* bufC = bufB + (size_t)NN * 256;
    _Float16* W1t  = (_Float16*)(bufC + (size_t)NN * 256);   // 256 x 128
    _Float16* Wfct = W1t  + (size_t)INDIM * HID4;            // 256 x 128
    _Float16* W2t  = Wfct + (size_t)INDIM * HID4;            // 256 x 512
    _Float16* W3t  = W2t  + (size_t)2 * HID4 * HID4;         // 256 x 256

    const int T = 256;
    dim3 gemmGrid(NN / 16, HID4 / 128);           // 3125 x 2 blocks, 8 waves each
    int spmmBlocks = (NE * 32 + T - 1) / T;       // one wave per edge
    int waveRowBlocks = ((NN * 32) + T - 1) / T;  // one wave per node row

    // 0) transpose+convert weights to f16 (tiny, L2-resident)
    convert_weight_t<<<(INDIM * HID4 + T - 1) / T, T, 0, stream>>>(W1,  W1t,  INDIM, HID4);
    convert_weight_t<<<(INDIM * HID4 + T - 1) / T, T, 0, stream>>>(Wfc, Wfct, INDIM, HID4);
    convert_weight_t<<<(2 * HID4 * HID4 + T - 1) / T, T, 0, stream>>>(W2, W2t, 2 * HID4, HID4);
    convert_weight_t<<<(HID4 * HID4 + T - 1) / T, T, 0, stream>>>(W3, W3t, HID4, HID4);

    // 1) degrees -> D^{-1/2} norms
    zero_kernel<<<(2 * NN + T - 1) / T, T, 0, stream>>>(outn, 2L * NN);
    degree_kernel<<<(NE + T - 1) / T, T, 0, stream>>>(src, dst, outn, inn, NE);
    norm_kernel<<<(NN + T - 1) / T, T, 0, stream>>>(outn, inn, NN);

    // 2) agg0 = segsum_e w_e * out_norm[src] * x[src] -> bufB (N x 128)
    zero_kernel<<<((long)NN * INDIM + T - 1) / T, T, 0, stream>>>(bufB, (long)NN * INDIM);
    spmm_kernel<<<spmmBlocks, T, 0, stream>>>(x, INDIM, src, dst, w, outn, bufB, NE);

    // 3) x1 = relu(in_norm * (agg0 @ W1)) -> x1f1[:, 0:256]
    gemm_wmma_f16<false, true, true><<<gemmGrid, T, 0, stream>>>(
        bufB, W1t, x1f1, INDIM, HID4, 512, nullptr, inn);
    // 4) f1_raw = x @ Wfc -> x1f1[:, 256:512]; then LayerNorm+ReLU in place
    gemm_wmma_f16<false, false, false><<<gemmGrid, T, 0, stream>>>(
        x, Wfct, x1f1 + HID4, INDIM, HID4, 512, nullptr, nullptr);
    layernorm_relu_kernel<<<waveRowBlocks, T, 0, stream>>>(x1f1 + HID4, 512, gamma, beta, NN);

    // 5) T2 = (x1f1 * out_norm) @ W2 -> bufB (N x 256)
    gemm_wmma_f16<true, false, false><<<gemmGrid, T, 0, stream>>>(
        x1f1, W2t, bufB, 2 * HID4, HID4, 256, outn, nullptr);

    // 6) x2 = relu(in_norm * segsum_e w_e * T2[src]) -> bufC
    zero_kernel<<<((long)NN * HID4 + T - 1) / T, T, 0, stream>>>(bufC, (long)NN * HID4);
    spmm_kernel<<<spmmBlocks, T, 0, stream>>>(bufB, HID4, src, dst, w, nullptr, bufC, NE);
    rowscale_relu_kernel<<<((long)NN * HID4 + T - 1) / T, T, 0, stream>>>(bufC, inn, NN, HID4);

    // 7) agg2 = segsum_e w_e * out_norm[src] * x2[src] -> bufB
    zero_kernel<<<((long)NN * HID4 + T - 1) / T, T, 0, stream>>>(bufB, (long)NN * HID4);
    spmm_kernel<<<spmmBlocks, T, 0, stream>>>(bufC, HID4, src, dst, w, outn, bufB, NE);

    // 8) x3 = relu(in_norm * (agg2 @ W3)) -> bufC
    gemm_wmma_f16<false, true, true><<<gemmGrid, T, 0, stream>>>(
        bufB, W3t, bufC, HID4, OUTDIM, 256, nullptr, inn);

    // 9) per-graph readout
    zero_kernel<<<(NG * OUTDIM + T - 1) / T, T, 0, stream>>>(out, (long)NG * OUTDIM);
    readout_kernel<<<(NN + 255) / 256, 256, 0, stream>>>(bufC, gid, out, NN);
}
